// ParticleNet_32873679684023
// MI455X (gfx1250) — compile-verified, matrix operand order checked by reading, not run on hardware
//
#include <hip/hip_runtime.h>

// ---------------------------------------------------------------------------
// ParticleNet-style message passing GNN, WMMA-accelerated for gfx1250.
//   edge MLP : 15 -> 128 -> 128 -> 128 -> 128   (tanh between)
//   scatter-add by dst                          (global_atomic_add_f32, asm)
//   node MLP : 134 -> 128 -> 128 -> 3           (tanh between)
// All 128-wide GEMM layers run on v_wmma_f32_16x16x32_f16 (f16 in, f32 acc).
// Edge kernel processes 32 edges / iteration (2 M-tiles, interleaved WMMA
// chains) with all weight fragments resident in VGPRs.
// ---------------------------------------------------------------------------

#define HIDDEN_C 128

typedef _Float16 v16h __attribute__((ext_vector_type(16)));
typedef _Float16 v8h  __attribute__((ext_vector_type(8)));
typedef float    v8f  __attribute__((ext_vector_type(8)));

union HFrag { v16h v; v8h h[2]; };

// tanh(x) = 1 - 2/(1+e^{2x}) using fast hardware rcp (v_rcp_f32), avoiding
// the IEEE div expansion (v_div_scale/v_div_fmas chains).
__device__ __forceinline__ float ftanh(float x) {
    x = fminf(fmaxf(x, -9.0f), 9.0f);           // v_med3
    float e = __expf(2.0f * x);                 // v_exp_f32
    float r = __builtin_amdgcn_rcpf(1.0f + e);  // v_rcp_f32
    return fmaf(-2.0f, r, 1.0f);                // v_fma_f32
}

__device__ __forceinline__ v8f wmma_f16(const v16h& a, const v16h& b, const v8f& c) {
    return __builtin_amdgcn_wmma_f32_16x16x32_f16(
        /*neg_a=*/false, a, /*neg_b=*/false, b,
        /*c_mod=*/(short)0, c, /*reuse_a=*/false, /*reuse_b=*/false);
}

// Native CDNA5 FP32 global atomic add, no-return form (STOREcnt-tracked).
__device__ __forceinline__ void atomic_add_f32(float* p, float v) {
    asm volatile("global_atomic_add_f32 %0, %1, off"
                 :: "v"(p), "v"(v) : "memory");
}

// ---------------------------------------------------------------------------
// Swizzle a row-major f32 weight matrix W[K x 128] into per-(ktile,ntile)
// WMMA B fragments: 512 halves per 32x16 tile, per-lane 16 halves contiguous
// so a wave fetches its fragment with two global_load_b128 per lane.
// K element for (e, half h): v=e>>1; lo=(v<4)? 2v : 16+2(v-4); K=lo+(e&1)+8h.
// ---------------------------------------------------------------------------
__global__ void swizzle_weights_kernel(const float* __restrict__ W, int Kreal,
                                       int ktiles, _Float16* __restrict__ dst) {
    int tid = blockIdx.x * blockDim.x + threadIdx.x;
    int tiles = ktiles * 8;
    if (tid >= tiles * 32) return;
    int tile = tid >> 5, lane = tid & 31;
    int ktile = tile >> 3, ntile = tile & 7;
    int n = ntile * 16 + (lane & 15);
    int h = lane >> 4;
    _Float16* out = dst + (size_t)tile * 512 + lane * 16;
#pragma unroll
    for (int e = 0; e < 16; ++e) {
        int v  = e >> 1;
        int lo = (v < 4) ? (v * 2) : (16 + (v - 4) * 2);
        int K  = ktile * 32 + lo + (e & 1) + h * 8;
        float val = (K < Kreal) ? W[(size_t)K * 128 + n] : 0.0f;
        out[e] = (_Float16)val;
    }
}

__device__ __forceinline__ void load_bfrag(HFrag& f, const _Float16* base,
                                           int ktile, int ntile, int lane) {
    const _Float16* p = base + ((size_t)(ktile * 8 + ntile) * 512) + lane * 16;
    f.h[0] = *(const v8h*)(p);
    f.h[1] = *(const v8h*)(p + 8);
}

__device__ __forceinline__ void load_afrag(HFrag& a, const _Float16* act,
                                           int rowBase, int stride, int kt,
                                           int m, int h) {
    const _Float16* p = act + (size_t)(rowBase + m) * stride + kt * 32 + h * 8;
    a.h[0] = *(const v8h*)(p);
    a.h[1] = *(const v8h*)(p + 16);
}

// ---------------------------------------------------------------------------
// Edge kernel: grid-stride over 32-edge blocks (2 M-tiles). 8 waves/WG,
// wave j owns HIDDEN columns [16j, 16j+16). 13 weight fragments stay in
// VGPRs for the whole kernel; 26 WMMAs per block iteration.
// ---------------------------------------------------------------------------
__global__ void __launch_bounds__(256)
edge_mlp_scatter_kernel(const float* __restrict__ x,
                        const int* __restrict__ eidx,
                        const float* __restrict__ eattr,
                        const _Float16* __restrict__ W0s,
                        const _Float16* __restrict__ W1s,
                        const _Float16* __restrict__ W2s,
                        const _Float16* __restrict__ W3s,
                        const float* __restrict__ b0,
                        const float* __restrict__ b1,
                        const float* __restrict__ b2,
                        const float* __restrict__ b3,
                        float* __restrict__ aggr, int E) {
    __shared__ __align__(16) _Float16 sIn[32 * 32];     // padded 15 -> 32
    __shared__ __align__(16) _Float16 sActA[32 * 128];
    __shared__ __align__(16) _Float16 sActB[32 * 128];
    __shared__ int sDst[32];

    const int tid  = threadIdx.x;
    const int lane = tid & 31;
    const int wav  = tid >> 5;          // column tile 0..7
    const int m    = lane & 15;
    const int h    = lane >> 4;
    const int ncol = wav * 16 + m;      // this lane's output column

    // --- resident weight fragments -----------------------------------------
    HFrag w0, w1[4], w2[4], w3[4];
    load_bfrag(w0, W0s, 0, wav, lane);
#pragma unroll
    for (int kt = 0; kt < 4; ++kt) {
        load_bfrag(w1[kt], W1s, kt, wav, lane);
        load_bfrag(w2[kt], W2s, kt, wav, lane);
        load_bfrag(w3[kt], W3s, kt, wav, lane);
    }
    const float bias0 = b0[ncol], bias1 = b1[ncol],
                bias2 = b2[ncol], bias3 = b3[ncol];

    const int nBlocks = (E + 31) >> 5;
    for (int eb = blockIdx.x; eb < nBlocks; eb += gridDim.x) {
        const int base = eb << 5;

        // prefetch next block's indices (global_prefetch_b8)
        {
            int nb = base + (gridDim.x << 5);
            if (nb < E) __builtin_prefetch(&eidx[nb], 0, 1);
        }

        // --- stage padded 32x32 input panel: [x_dst(6) | x_src(6) | ea(3) | 0]
#pragma unroll
        for (int idx = tid; idx < 32 * 32; idx += 256) {
            int row = idx >> 5, col = idx & 31;
            int e = base + row;
            float v = 0.0f;
            if (e < E) {
                if (col < 6)        v = x[(size_t)eidx[E + e] * 6 + col];
                else if (col < 12)  v = x[(size_t)eidx[e] * 6 + (col - 6)];
                else if (col < 15)  v = eattr[(size_t)e * 3 + (col - 12)];
            }
            sIn[idx] = (_Float16)v;
        }
        if (tid < 32) {
            int e = base + tid;
            sDst[tid] = (e < E) ? eidx[E + e] : -1;
        }
        __syncthreads();

        HFrag a;
        v8f c0, c1;

        // --- layer 0: 15(->32) x 128, two M-tiles --------------------------
        c0 = (v8f){}; c1 = (v8f){};
        load_afrag(a, sIn, 0, 32, 0, m, h);
        c0 = wmma_f16(a.v, w0.v, c0);
        load_afrag(a, sIn, 16, 32, 0, m, h);
        c1 = wmma_f16(a.v, w0.v, c1);
#pragma unroll
        for (int r = 0; r < 8; ++r) {
            int rr = r + 8 * h;
            sActA[rr * 128 + ncol]        = (_Float16)ftanh(c0[r] + bias0);
            sActA[(16 + rr) * 128 + ncol] = (_Float16)ftanh(c1[r] + bias0);
        }
        __syncthreads();

        // --- layer 1: actA -> actB -----------------------------------------
        c0 = (v8f){}; c1 = (v8f){};
#pragma unroll
        for (int kt = 0; kt < 4; ++kt) {
            load_afrag(a, sActA, 0, 128, kt, m, h);
            c0 = wmma_f16(a.v, w1[kt].v, c0);
            load_afrag(a, sActA, 16, 128, kt, m, h);
            c1 = wmma_f16(a.v, w1[kt].v, c1);
        }
#pragma unroll
        for (int r = 0; r < 8; ++r) {
            int rr = r + 8 * h;
            sActB[rr * 128 + ncol]        = (_Float16)ftanh(c0[r] + bias1);
            sActB[(16 + rr) * 128 + ncol] = (_Float16)ftanh(c1[r] + bias1);
        }
        __syncthreads();

        // --- layer 2: actB -> actA -----------------------------------------
        c0 = (v8f){}; c1 = (v8f){};
#pragma unroll
        for (int kt = 0; kt < 4; ++kt) {
            load_afrag(a, sActB, 0, 128, kt, m, h);
            c0 = wmma_f16(a.v, w2[kt].v, c0);
            load_afrag(a, sActB, 16, 128, kt, m, h);
            c1 = wmma_f16(a.v, w2[kt].v, c1);
        }
#pragma unroll
        for (int r = 0; r < 8; ++r) {
            int rr = r + 8 * h;
            sActA[rr * 128 + ncol]        = (_Float16)ftanh(c0[r] + bias2);
            sActA[(16 + rr) * 128 + ncol] = (_Float16)ftanh(c1[r] + bias2);
        }
        __syncthreads();

        // --- layer 3 (no tanh) -> scatter-add to aggr ----------------------
        c0 = (v8f){}; c1 = (v8f){};
#pragma unroll
        for (int kt = 0; kt < 4; ++kt) {
            load_afrag(a, sActA, 0, 128, kt, m, h);
            c0 = wmma_f16(a.v, w3[kt].v, c0);
            load_afrag(a, sActA, 16, 128, kt, m, h);
            c1 = wmma_f16(a.v, w3[kt].v, c1);
        }
#pragma unroll
        for (int r = 0; r < 8; ++r) {
            int rr = r + 8 * h;
            int d0 = sDst[rr];
            if (d0 >= 0)
                atomic_add_f32(&aggr[(size_t)d0 * HIDDEN_C + ncol], c0[r] + bias3);
            int d1 = sDst[16 + rr];
            if (d1 >= 0)
                atomic_add_f32(&aggr[(size_t)d1 * HIDDEN_C + ncol], c1[r] + bias3);
        }
        __syncthreads();                     // protect sIn/sActA for next block
    }
}

// ---------------------------------------------------------------------------
// Node kernel: [x(6) | aggr(128)] padded to 160 -> 128 -> 128 -> 3.
// ---------------------------------------------------------------------------
__global__ void __launch_bounds__(256)
node_mlp_kernel(const float* __restrict__ x,
                const float* __restrict__ aggr,
                const _Float16* __restrict__ U0s,
                const _Float16* __restrict__ U1s,
                const float* __restrict__ uW2,   // [128 x 3] row-major f32
                const float* __restrict__ ub0,
                const float* __restrict__ ub1,
                const float* __restrict__ ub2,
                float* __restrict__ out, int N) {
    __shared__ __align__(16) _Float16 sIn[16 * 160];    // 134 -> 160 padded
    __shared__ __align__(16) _Float16 sAct[16 * 128];
    __shared__ __align__(16) float    sFin[16 * 128];

    const int tid  = threadIdx.x;
    const int lane = tid & 31;
    const int wav  = tid >> 5;
    const int m    = lane & 15;
    const int h    = lane >> 4;
    const int ncol = wav * 16 + m;

    HFrag u0[5], u1[4];
#pragma unroll
    for (int kt = 0; kt < 5; ++kt) load_bfrag(u0[kt], U0s, kt, wav, lane);
#pragma unroll
    for (int kt = 0; kt < 4; ++kt) load_bfrag(u1[kt], U1s, kt, wav, lane);
    const float bias0 = ub0[ncol], bias1 = ub1[ncol];

    const int nBlocks = (N + 15) >> 4;
    for (int nb = blockIdx.x; nb < nBlocks; nb += gridDim.x) {
        const int base = nb << 4;

        // --- stage padded 16x160 input: [x(6) | aggr(128) | 0] --------------
#pragma unroll
        for (int idx = tid; idx < 16 * 160; idx += 256) {
            int row = idx / 160, col = idx - row * 160;
            int node = base + row;
            float v = 0.0f;
            if (node < N) {
                if (col < 6)         v = x[(size_t)node * 6 + col];
                else if (col < 134)  v = aggr[(size_t)node * 128 + (col - 6)];
            }
            sIn[idx] = (_Float16)v;
        }
        __syncthreads();

        // --- layer 0: 134(->160) x 128 -------------------------------------
        HFrag a;
        v8f c = {};
#pragma unroll
        for (int kt = 0; kt < 5; ++kt) {
            load_afrag(a, sIn, 0, 160, kt, m, h);
            c = wmma_f16(a.v, u0[kt].v, c);
        }
#pragma unroll
        for (int r = 0; r < 8; ++r)
            sAct[(r + 8 * h) * 128 + ncol] = (_Float16)ftanh(c[r] + bias0);
        __syncthreads();

        // --- layer 1: 128 x 128 --------------------------------------------
        c = (v8f){};
#pragma unroll
        for (int kt = 0; kt < 4; ++kt) {
            load_afrag(a, sAct, 0, 128, kt, m, h);
            c = wmma_f16(a.v, u1[kt].v, c);
        }
#pragma unroll
        for (int r = 0; r < 8; ++r)
            sFin[(r + 8 * h) * 128 + ncol] = ftanh(c[r] + bias1);
        __syncthreads();

        // --- layer 2: 128 -> 3 (scalar f32 dot products) -------------------
        if (tid < 48) {
            int row = tid / 3, o = tid - row * 3;
            int node = base + row;
            if (node < N) {
                float acc = ub2[o];
                const float* f = &sFin[row * 128];
#pragma unroll 8
                for (int k = 0; k < 128; ++k)
                    acc = fmaf(f[k], uW2[k * 3 + o], acc);
                out[(size_t)node * 3 + o] = acc;
            }
        }
        __syncthreads();                     // protect sIn/sFin for next block
    }
}

// ---------------------------------------------------------------------------
// Host launcher
// ---------------------------------------------------------------------------
extern "C" void kernel_launch(void* const* d_in, const int* in_sizes, int n_in,
                              void* d_out, int out_size, void* d_ws, size_t ws_size,
                              hipStream_t stream) {
    const float* x     = (const float*)d_in[0];
    const int*   eidx  = (const int*)d_in[1];
    const float* eattr = (const float*)d_in[2];
    const float* mW0 = (const float*)d_in[3];  const float* mB0 = (const float*)d_in[4];
    const float* mW1 = (const float*)d_in[5];  const float* mB1 = (const float*)d_in[6];
    const float* mW2 = (const float*)d_in[7];  const float* mB2 = (const float*)d_in[8];
    const float* mW3 = (const float*)d_in[9];  const float* mB3 = (const float*)d_in[10];
    const float* uW0 = (const float*)d_in[11]; const float* uB0 = (const float*)d_in[12];
    const float* uW1 = (const float*)d_in[13]; const float* uB1 = (const float*)d_in[14];
    const float* uW2 = (const float*)d_in[15]; const float* uB2 = (const float*)d_in[16];

    const int N = in_sizes[0] / 6;      // 50000
    const int E = in_sizes[1] / 2;      // 1600000
    float* out  = (float*)d_out;

    // workspace layout
    char*  ws   = (char*)d_ws;
    float* aggr = (float*)ws;
    size_t off  = (size_t)N * HIDDEN_C * sizeof(float);
    _Float16* w0s = (_Float16*)(ws + off); off += (size_t)1 * 8 * 512 * 2;
    _Float16* w1s = (_Float16*)(ws + off); off += (size_t)4 * 8 * 512 * 2;
    _Float16* w2s = (_Float16*)(ws + off); off += (size_t)4 * 8 * 512 * 2;
    _Float16* w3s = (_Float16*)(ws + off); off += (size_t)4 * 8 * 512 * 2;
    _Float16* u0s = (_Float16*)(ws + off); off += (size_t)5 * 8 * 512 * 2;
    _Float16* u1s = (_Float16*)(ws + off); off += (size_t)4 * 8 * 512 * 2;

    hipMemsetAsync(aggr, 0, (size_t)N * HIDDEN_C * sizeof(float), stream);

    auto swzGrid = [](int ktiles) { return (ktiles * 8 * 32 + 255) / 256; };
    swizzle_weights_kernel<<<swzGrid(1), 256, 0, stream>>>(mW0,  15, 1, w0s);
    swizzle_weights_kernel<<<swzGrid(4), 256, 0, stream>>>(mW1, 128, 4, w1s);
    swizzle_weights_kernel<<<swzGrid(4), 256, 0, stream>>>(mW2, 128, 4, w2s);
    swizzle_weights_kernel<<<swzGrid(4), 256, 0, stream>>>(mW3, 128, 4, w3s);
    swizzle_weights_kernel<<<swzGrid(5), 256, 0, stream>>>(uW0, 134, 5, u0s);
    swizzle_weights_kernel<<<swzGrid(4), 256, 0, stream>>>(uW1, 128, 4, u1s);

    edge_mlp_scatter_kernel<<<4096, 256, 0, stream>>>(
        x, eidx, eattr, w0s, w1s, w2s, w3s, mB0, mB1, mB2, mB3, aggr, E);

    node_mlp_kernel<<<1024, 256, 0, stream>>>(
        x, aggr, u0s, u1s, uW2, uB0, uB1, uB2, out, N);
}
